// AbsorberPathAggregator_34797825032825
// MI455X (gfx1250) — matrix operand order repaired and checked by reading, not compiled, not run on hardware
//
#include <hip/hip_runtime.h>
#include <math.h>

typedef __attribute__((ext_vector_type(16))) _Float16 v16h;
typedef __attribute__((ext_vector_type(8)))  _Float16 v8h;
typedef __attribute__((ext_vector_type(8)))  float    v8f;

#define CUTOFF_R 5.0f
#define ATOM_DIM 128
#define N_E      80
#define SCAT     64
#define N_B      16
#define P_MAX    8000
#define PI_F     3.14159265358979323846f

// ---------------- workspace layout ----------------
// f16 region (units: halves)
#define H_GM1T  ((size_t)0)                    // 128 x 384 (K padded 353->384)
#define H_GM2T  (H_GM1T + 128*384)             // 128 x 128
#define H_GM3T  (H_GM2T + 128*128)             // 64  x 128
#define H_PE1AT (H_GM3T + 64*128)              // 64  x 32
#define H_PE1BT (H_PE1AT + 64*32)              // 64  x 32
#define H_PE2T  (H_PE1BT + 64*32)              // 64  x 64
#define H_PE3T  (H_PE2T + 64*64)               // 64  x 64
#define H_OP1T  (H_PE3T + 64*64)               // 128 x 64
#define H_OP2T  (H_OP1T + 128*64)              // 64  x 128
#define H_TOTAL (H_OP2T + 64*128)              // 102400 halves
#define WSF_BYTE_OFF (H_TOTAL * 2)             // 204800 bytes, 256B aligned
// f32 region (units: floats)
#define F_EE    ((size_t)0)                    // 80*64  : e_feat@W1c + b1
#define F_GG    (F_EE + N_E*SCAT)              // P*64   : g_geom (linear out)
#define F_PEPRE (F_GG + (size_t)P_MAX*SCAT)    // P*64   : ej@W1a + ek@W1b
#define F_CW    (F_PEPRE + (size_t)P_MAX*SCAT) // P      : cutoff weights
#define F_AGG   (F_CW + P_MAX)                 // B*NE*SCAT accumulator
#define F_NORM  (F_AGG + (size_t)N_B*N_E*SCAT) // B

// ---------------- WMMA / async helpers ----------------
__device__ __forceinline__ v8f wmma32(v16h a, v16h b, v8f c) {
  return __builtin_amdgcn_wmma_f32_16x16x32_f16(false, a, false, b, (short)0, c,
                                                false, false);
}
// A fragment: 16x32 f16 tile, row m = lane&15; chunks at k0+16*hi and k0+16*hi+8
__device__ __forceinline__ v16h frag_a(const _Float16* s, int stride, int m,
                                       int k0, int hi) {
  const _Float16* p = s + m * stride + k0 + hi * 16;
  v8h lo = *(const v8h*)p;
  v8h hi8 = *(const v8h*)(p + 8);
  v16h a;
#pragma unroll
  for (int i = 0; i < 8; ++i) { a[i] = lo[i]; a[8 + i] = hi8[i]; }
  return a;
}
// B fragment from transposed weights wT[N][K]: contiguous 16 halves of row n
__device__ __forceinline__ v16h frag_b(const _Float16* wT, int K, int n, int k0,
                                       int hi) {
  return *(const v16h*)(wT + (size_t)n * K + k0 + hi * 16);
}
__device__ __forceinline__ float silu(float x) {
  return x / (1.0f + __expf(-x));
}
__device__ __forceinline__ float cutfn(float r) {
  return (r < CUTOFF_R) ? 0.5f * (__cosf(PI_F * r / CUTOFF_R) + 1.0f) : 0.0f;
}
// CDNA5 async global->LDS copy (per-lane 16B), tracked by ASYNCcnt
__device__ __forceinline__ void async_g2l_b128(void* lds_ptr, const void* gptr) {
  unsigned lds_off = (unsigned)(unsigned long long)(uintptr_t)lds_ptr;
  unsigned long long ga = (unsigned long long)(uintptr_t)gptr;
  asm volatile("global_load_async_to_lds_b128 %0, %1, off"
               :: "v"(lds_off), "v"(ga) : "memory");
}
__device__ __forceinline__ void wait_async0() {
  asm volatile("s_wait_asynccnt 0" ::: "memory");
}
__device__ __forceinline__ void wait_ds0() {
  asm volatile("s_wait_dscnt 0" ::: "memory");
}

// ---------------- kernel 1: weight prep / transpose / zero ----------------
__global__ void k_prep(const float* gm_w1, const float* gm_w2, const float* gm_w3,
                       const float* pe_w1, const float* pe_b1, const float* pe_w2,
                       const float* pe_w3, const float* op_w1, const float* op_w2,
                       const float* e_feat, _Float16* wsH, float* wsF) {
  int tid = blockIdx.x * blockDim.x + threadIdx.x;
  int nthr = gridDim.x * blockDim.x;
  for (int i = tid; i < 128 * 384; i += nthr) {
    int n = i / 384, k = i - n * 384;
    wsH[H_GM1T + i] = (_Float16)(k < 353 ? gm_w1[k * 128 + n] : 0.0f);
  }
  for (int i = tid; i < 128 * 128; i += nthr) {
    int n = i >> 7, k = i & 127;
    wsH[H_GM2T + i] = (_Float16)gm_w2[k * 128 + n];
  }
  for (int i = tid; i < 64 * 128; i += nthr) {
    int n = i >> 7, k = i & 127;
    wsH[H_GM3T + i] = (_Float16)gm_w3[k * 64 + n];
  }
  for (int i = tid; i < 64 * 32; i += nthr) {
    int n = i >> 5, k = i & 31;
    wsH[H_PE1AT + i] = (_Float16)pe_w1[k * 64 + n];
    wsH[H_PE1BT + i] = (_Float16)pe_w1[(k + 32) * 64 + n];
  }
  for (int i = tid; i < 64 * 64; i += nthr) {
    int n = i >> 6, k = i & 63;
    wsH[H_PE2T + i] = (_Float16)pe_w2[k * 64 + n];
    wsH[H_PE3T + i] = (_Float16)pe_w3[k * 64 + n];
  }
  for (int i = tid; i < 128 * 64; i += nthr) {
    int n = i >> 6, k = i & 63;
    wsH[H_OP1T + i] = (_Float16)op_w1[k * 128 + n];
  }
  for (int i = tid; i < 64 * 128; i += nthr) {
    int n = i >> 7, k = i & 127;
    wsH[H_OP2T + i] = (_Float16)op_w2[k * 64 + n];
  }
  // ee_pre = e_feat @ W1c + b1  (80 x 64, tiny)
  for (int i = tid; i < N_E * 64; i += nthr) {
    int e = i >> 6, n = i & 63;
    float acc = pe_b1[n];
    for (int k = 0; k < 32; ++k) acc += e_feat[e * 32 + k] * pe_w1[(64 + k) * 64 + n];
    wsF[F_EE + i] = acc;
  }
  for (int i = tid; i < N_B * N_E * SCAT; i += nthr) wsF[F_AGG + i] = 0.0f;
  for (int i = tid; i < N_B; i += nthr) wsF[F_NORM + i] = 0.0f;
}

// ---------------- kernel 2: geometry MLP3 + pe_pre + cw (16 paths/block) -----
__global__ void __launch_bounds__(128) k_geom(
    const float* h_flat, const int* z_flat, const float* z_emb, const int* path_j,
    const int* path_k, const float* r0j, const float* r0k, const float* rjk,
    const float* cosang, const int* path_batch, const float* gm_b1,
    const float* gm_b2, const float* gm_b3, const _Float16* wsH, float* wsF,
    int P) {
  __shared__ __align__(16) _Float16 sIn[16 * 392];  // K padded to 384, stride 392
  __shared__ __align__(16) _Float16 sEj[16 * 32];
  __shared__ __align__(16) _Float16 sEk[16 * 32];
  __shared__ __align__(16) _Float16 sH1[16 * 128];
  __shared__ __align__(16) _Float16 sH2[16 * 128];
  const int p0 = blockIdx.x * 16;
  const int t = threadIdx.x;
  // stage geom_in rows (hj | hk | rbf0j | rbf0k | rbfjk | cos | zero-pad)
  for (int i = t; i < 16 * 384; i += 128) {
    int row = i / 384, c = i - row * 384;
    int p = min(p0 + row, P - 1);
    float v;
    if (c < 128)      v = h_flat[path_j[p] * ATOM_DIM + c];
    else if (c < 256) v = h_flat[path_k[p] * ATOM_DIM + (c - 128)];
    else if (c < 352) {
      int seg = (c - 256) >> 5, idx = (c - 256) & 31;
      float r = (seg == 0) ? r0j[p] : ((seg == 1) ? r0k[p] : rjk[p]);
      r = fminf(r, CUTOFF_R);
      float w = CUTOFF_R / 31.0f;
      float d = (r - w * (float)idx) / w;
      v = __expf(-0.5f * d * d);
    } else if (c == 352) v = cosang[p];
    else                 v = 0.0f;
    sIn[row * 392 + c] = (_Float16)v;
  }
  for (int i = t; i < 16 * 32; i += 128) {
    int row = i >> 5, c = i & 31;
    int p = min(p0 + row, P - 1);
    sEj[i] = (_Float16)z_emb[z_flat[path_j[p]] * 32 + c];
    sEk[i] = (_Float16)z_emb[z_flat[path_k[p]] * 32 + c];
  }
  __syncthreads();
  const int wave = t >> 5, lane = t & 31, hi = lane >> 4, nc = lane & 15;
  // pe_pre = ej@W1a + ek@W1b : one 16x64 tile, wave handles ntile = wave
  {
    int nt = wave;
    v8f c = {0.f, 0.f, 0.f, 0.f, 0.f, 0.f, 0.f, 0.f};
    v16h aj = frag_a(sEj, 32, nc, 0, hi);
    v16h ak = frag_a(sEk, 32, nc, 0, hi);
    c = wmma32(aj, frag_b(wsH + H_PE1AT, 32, nt * 16 + nc, 0, hi), c);
    c = wmma32(ak, frag_b(wsH + H_PE1BT, 32, nt * 16 + nc, 0, hi), c);
    int col = nt * 16 + nc;
#pragma unroll
    for (int r = 0; r < 8; ++r)
      wsF[F_PEPRE + (size_t)(p0 + r + 8 * hi) * 64 + col] = c[r];
  }
  // layer 1: 16x384 @ 384x128
  for (int ntI = 0; ntI < 2; ++ntI) {
    int nt = wave + ntI * 4;
    v8f c = {0.f, 0.f, 0.f, 0.f, 0.f, 0.f, 0.f, 0.f};
    for (int kf = 0; kf < 12; ++kf) {
      v16h a = frag_a(sIn, 392, nc, kf * 32, hi);
      v16h b = frag_b(wsH + H_GM1T, 384, nt * 16 + nc, kf * 32, hi);
      c = wmma32(a, b, c);
    }
    int col = nt * 16 + nc;
    float bias = gm_b1[col];
#pragma unroll
    for (int r = 0; r < 8; ++r)
      sH1[(r + 8 * hi) * 128 + col] = (_Float16)silu(c[r] + bias);
  }
  __syncthreads();
  // layer 2: 16x128 @ 128x128
  for (int ntI = 0; ntI < 2; ++ntI) {
    int nt = wave + ntI * 4;
    v8f c = {0.f, 0.f, 0.f, 0.f, 0.f, 0.f, 0.f, 0.f};
    for (int kf = 0; kf < 4; ++kf) {
      v16h a = frag_a(sH1, 128, nc, kf * 32, hi);
      v16h b = frag_b(wsH + H_GM2T, 128, nt * 16 + nc, kf * 32, hi);
      c = wmma32(a, b, c);
    }
    int col = nt * 16 + nc;
    float bias = gm_b2[col];
#pragma unroll
    for (int r = 0; r < 8; ++r)
      sH2[(r + 8 * hi) * 128 + col] = (_Float16)silu(c[r] + bias);
  }
  __syncthreads();
  // layer 3: 16x128 @ 128x64 -> g_geom (linear)
  {
    int nt = wave;
    v8f c = {0.f, 0.f, 0.f, 0.f, 0.f, 0.f, 0.f, 0.f};
    for (int kf = 0; kf < 4; ++kf) {
      v16h a = frag_a(sH2, 128, nc, kf * 32, hi);
      v16h b = frag_b(wsH + H_GM3T, 128, nt * 16 + nc, kf * 32, hi);
      c = wmma32(a, b, c);
    }
    int col = nt * 16 + nc;
    float bias = gm_b3[col];
#pragma unroll
    for (int r = 0; r < 8; ++r)
      wsF[F_GG + (size_t)(p0 + r + 8 * hi) * 64 + col] = c[r] + bias;
  }
  // cutoff weights + norm
  if (t < 16) {
    int p = p0 + t;
    if (p < P) {
      float cw = cutfn(r0j[p]) * cutfn(r0k[p]) * cutfn(rjk[p]);
      wsF[F_CW + p] = cw;
      atomicAdd(&wsF[F_NORM + path_batch[p]], cw);
    }
  }
}

// ---------------- kernel 3: pair-element MLP (dominant GEMMs) ----------------
__global__ void __launch_bounds__(256) k_elem(const int* path_batch,
                                              const float* pe_b2,
                                              const float* pe_b3,
                                              const _Float16* wsH, float* wsF,
                                              int P) {
  __shared__ __align__(16) float sPe[16 * 64];
  __shared__ __align__(16) float sEe[N_E * 64];
  __shared__ __align__(16) float sGcw[16 * 64];
  __shared__ float sAcc[N_E * 64];
  __shared__ __align__(16) _Float16 sHst[8][16 * 64];
  __shared__ __align__(16) _Float16 sW2[64 * 64];
  __shared__ __align__(16) _Float16 sW3[64 * 64];
  __shared__ int sBatch[16];
  const int p0 = blockIdx.x * 16;
  const int t = threadIdx.x;
  // --- async global->LDS staging of block-invariant data (ASYNCcnt path) ---
#pragma unroll
  for (int off = t * 16; off < 64 * 64 * 2; off += 256 * 16) {
    async_g2l_b128((char*)sW2 + off, (const char*)(wsH + H_PE2T) + off);
    async_g2l_b128((char*)sW3 + off, (const char*)(wsH + H_PE3T) + off);
  }
  for (int off = t * 16; off < N_E * 64 * 4; off += 256 * 16)
    async_g2l_b128((char*)sEe + off, (const char*)(wsF + F_EE) + off);
  for (int off = t * 16; off < 16 * 64 * 4; off += 256 * 16)
    async_g2l_b128((char*)sPe + off,
                   (const char*)(wsF + F_PEPRE + (size_t)p0 * 64) + off);
  // --- regular staging (needs arithmetic) ---
  for (int i = t; i < 16 * 64; i += 256) {
    int pl = i >> 6;
    sGcw[i] = wsF[F_CW + p0 + pl] * wsF[F_GG + (size_t)p0 * 64 + i];
  }
  for (int i = t; i < N_E * 64; i += 256) sAcc[i] = 0.0f;
  if (t < 16) sBatch[t] = path_batch[min(p0 + t, P - 1)];
  wait_async0();
  __syncthreads();
  const int prim = sBatch[0];
  const int wave = t >> 5, lane = t & 31, hi = lane >> 4, nc = lane & 15;
  _Float16* myH = &sHst[wave][0];
  float* agg = wsF + F_AGG;
  for (int jobi = 0; jobi < 10; ++jobi) {
    int job = wave + jobi * 8;     // 80 jobs: 16 paths x 5 e-tiles
    int pl = job / 5, et = job - pl * 5;
    // layer 1 (factored): h1 = silu(pe_pre[p] + ee_pre[e])
    for (int i = lane; i < 16 * 64; i += 32) {
      int row = i >> 6, col = i & 63;
      float x = sPe[pl * 64 + col] + sEe[(et * 16 + row) * 64 + col];
      myH[row * 64 + col] = (_Float16)silu(x);
    }
    wait_ds0();   // wave-local LDS ordering (myH is per-wave)
    // layer 2: 16x64 @ 64x64 (weights resident in LDS)
    v16h a0 = frag_a(myH, 64, nc, 0, hi);
    v16h a1 = frag_a(myH, 64, nc, 32, hi);
    v8f c2[4];
#pragma unroll
    for (int nt = 0; nt < 4; ++nt) {
      v8f c = {0.f, 0.f, 0.f, 0.f, 0.f, 0.f, 0.f, 0.f};
      c = wmma32(a0, frag_b(sW2, 64, nt * 16 + nc, 0, hi), c);
      c = wmma32(a1, frag_b(sW2, 64, nt * 16 + nc, 32, hi), c);
      c2[nt] = c;
    }
    wait_ds0();
#pragma unroll
    for (int nt = 0; nt < 4; ++nt) {
      int col = nt * 16 + nc;
      float bias = pe_b2[col];
#pragma unroll
      for (int r = 0; r < 8; ++r)
        myH[(r + 8 * hi) * 64 + col] = (_Float16)silu(c2[nt][r] + bias);
    }
    wait_ds0();
    // layer 3 + fused contrib accumulation
    a0 = frag_a(myH, 64, nc, 0, hi);
    a1 = frag_a(myH, 64, nc, 32, hi);
    int bpl = sBatch[pl];
#pragma unroll
    for (int nt = 0; nt < 4; ++nt) {
      v8f c = {0.f, 0.f, 0.f, 0.f, 0.f, 0.f, 0.f, 0.f};
      c = wmma32(a0, frag_b(sW3, 64, nt * 16 + nc, 0, hi), c);
      c = wmma32(a1, frag_b(sW3, 64, nt * 16 + nc, 32, hi), c);
      int col = nt * 16 + nc;
      float bias = pe_b3[col];
      float gcw = sGcw[pl * 64 + col];  // cw[p] * g_geom[p, col]
#pragma unroll
      for (int r = 0; r < 8; ++r) {
        int e = et * 16 + r + 8 * hi;
        float val = (c[r] + bias) * gcw;
        if (bpl == prim)
          atomicAdd(&sAcc[e * 64 + col], val);       // ds_add_f32
        else
          atomicAdd(&agg[((size_t)bpl * N_E + e) * 64 + col], val);
      }
    }
    wait_ds0();
  }
  __syncthreads();
  for (int i = t; i < N_E * 64; i += 256)
    atomicAdd(&agg[(size_t)prim * N_E * 64 + i], sAcc[i]);
}

// ---------------- kernel 4: normalize + output MLP2 ----------------
__global__ void __launch_bounds__(128) k_out(const float* op_b1,
                                             const float* op_b2,
                                             const _Float16* wsH,
                                             const float* wsF, float* out) {
  __shared__ __align__(16) _Float16 sA[16 * 64];
  __shared__ __align__(16) _Float16 sH[16 * 128];
  const int tile = blockIdx.x;              // 80 tiles of 16 rows; 5 tiles/batch
  const int b = tile / 5, e0 = (tile - b * 5) * 16;
  const int t = threadIdx.x;
  float inv = 1.0f / fmaxf(wsF[F_NORM + b], 1e-8f);
  for (int i = t; i < 16 * 64; i += 128) {
    int row = i >> 6, col = i & 63;
    sA[i] = (_Float16)(wsF[F_AGG + ((size_t)b * N_E + (e0 + row)) * 64 + col] * inv);
  }
  __syncthreads();
  const int wave = t >> 5, lane = t & 31, hi = lane >> 4, nc = lane & 15;
  v16h a0 = frag_a(sA, 64, nc, 0, hi);
  v16h a1 = frag_a(sA, 64, nc, 32, hi);
  for (int ntI = 0; ntI < 2; ++ntI) {
    int nt = wave + ntI * 4;
    v8f c = {0.f, 0.f, 0.f, 0.f, 0.f, 0.f, 0.f, 0.f};
    c = wmma32(a0, frag_b(wsH + H_OP1T, 64, nt * 16 + nc, 0, hi), c);
    c = wmma32(a1, frag_b(wsH + H_OP1T, 64, nt * 16 + nc, 32, hi), c);
    int col = nt * 16 + nc;
    float bias = op_b1[col];
#pragma unroll
    for (int r = 0; r < 8; ++r)
      sH[(r + 8 * hi) * 128 + col] = (_Float16)silu(c[r] + bias);
  }
  __syncthreads();
  {
    int nt = wave;
    v8f c = {0.f, 0.f, 0.f, 0.f, 0.f, 0.f, 0.f, 0.f};
    for (int kf = 0; kf < 4; ++kf) {
      v16h a = frag_a(sH, 128, nc, kf * 32, hi);
      v16h bb = frag_b(wsH + H_OP2T, 128, nt * 16 + nc, kf * 32, hi);
      c = wmma32(a, bb, c);
    }
    int col = nt * 16 + nc;
    float bias = op_b2[col];
#pragma unroll
    for (int r = 0; r < 8; ++r) {
      int g = tile * 16 + r + 8 * hi;
      out[(size_t)g * 64 + col] = c[r] + bias;
    }
  }
}

extern "C" void kernel_launch(void* const* d_in, const int* in_sizes, int n_in,
                              void* d_out, int out_size, void* d_ws,
                              size_t ws_size, hipStream_t stream) {
  const float* h_flat   = (const float*)d_in[0];
  const int*   z_flat   = (const int*)d_in[1];
  const float* e_feat   = (const float*)d_in[2];
  const int*   path_j   = (const int*)d_in[3];
  const int*   path_k   = (const int*)d_in[4];
  const float* r0j      = (const float*)d_in[5];
  const float* r0k      = (const float*)d_in[6];
  const float* rjk      = (const float*)d_in[7];
  const float* cosang   = (const float*)d_in[8];
  const int*   path_bat = (const int*)d_in[9];
  const float* z_emb    = (const float*)d_in[11];
  const float* pe_w1 = (const float*)d_in[12]; const float* pe_b1 = (const float*)d_in[13];
  const float* pe_w2 = (const float*)d_in[14]; const float* pe_b2 = (const float*)d_in[15];
  const float* pe_w3 = (const float*)d_in[16]; const float* pe_b3 = (const float*)d_in[17];
  const float* gm_w1 = (const float*)d_in[18]; const float* gm_b1 = (const float*)d_in[19];
  const float* gm_w2 = (const float*)d_in[20]; const float* gm_b2 = (const float*)d_in[21];
  const float* gm_w3 = (const float*)d_in[22]; const float* gm_b3 = (const float*)d_in[23];
  const float* op_w1 = (const float*)d_in[24]; const float* op_b1 = (const float*)d_in[25];
  const float* op_w2 = (const float*)d_in[26]; const float* op_b2 = (const float*)d_in[27];
  int P = in_sizes[3];
  if (P > P_MAX) P = P_MAX;
  _Float16* wsH = (_Float16*)d_ws;
  float* wsF = (float*)((char*)d_ws + WSF_BYTE_OFF);

  k_prep<<<64, 256, 0, stream>>>(gm_w1, gm_w2, gm_w3, pe_w1, pe_b1, pe_w2, pe_w3,
                                 op_w1, op_w2, e_feat, wsH, wsF);
  int nblk = (P + 15) / 16;
  k_geom<<<nblk, 128, 0, stream>>>(h_flat, z_flat, z_emb, path_j, path_k, r0j,
                                   r0k, rjk, cosang, path_bat, gm_b1, gm_b2,
                                   gm_b3, wsH, wsF, P);
  k_elem<<<nblk, 256, 0, stream>>>(path_bat, pe_b2, pe_b3, wsH, wsF, P);
  k_out<<<(N_B * N_E) / 16, 128, 0, stream>>>(op_b1, op_b2, wsH, wsF,
                                              (float*)d_out);
  (void)n_in; (void)out_size; (void)ws_size;
}
